// GNNModel_6158983102972
// MI455X (gfx1250) — matrix operand order repaired and checked by reading, not compiled, not run on hardware
//
#include <hip/hip_runtime.h>
#include <hip/hip_bf16.h>
#include <float.h>

// Problem constants (match reference)
#define NNODES   40000
#define NEDGES   640000
#define E2       (NEDGES + NNODES)   // with self loops
#define IN_DIM   5
#define EDGE_DIM 11
#define HID      128
#define NH1      4
#define D1       (NH1 * HID)         // 512

typedef __attribute__((ext_vector_type(16))) _Float16 v16h;
typedef __attribute__((ext_vector_type(8)))  float    v8f;
typedef __attribute__((ext_vector_type(4))) unsigned int tdm_v4u;
typedef __attribute__((ext_vector_type(8))) int          tdm_v8i;
typedef __attribute__((ext_vector_type(4))) int          tdm_v4i;

// float atomic max via sign-split int/uint atomics (amax initialized to -FLT_MAX)
__device__ __forceinline__ void atomicMaxF32(float* addr, float v) {
  if (v >= 0.0f) atomicMax((int*)addr, __float_as_int(v));
  else           atomicMin((unsigned int*)addr, __float_as_uint(v));
}

__global__ void k_fill(float* p, float v, int n) {
  int i = blockIdx.x * blockDim.x + threadIdx.x;
  if (i < n) p[i] = v;
}

// mean of edge_attr rows (accumulate; divided later in k_prep_loop)
__global__ void k_eamean(const float* __restrict__ ea, float* __restrict__ mean) {
  float loc[EDGE_DIM];
#pragma unroll
  for (int d = 0; d < EDGE_DIM; ++d) loc[d] = 0.f;
  int stride = gridDim.x * blockDim.x;
  for (int e = blockIdx.x * blockDim.x + threadIdx.x; e < NEDGES; e += stride) {
#pragma unroll
    for (int d = 0; d < EDGE_DIM; ++d) loc[d] += ea[(size_t)e * EDGE_DIM + d];
  }
  __shared__ float s[EDGE_DIM];
  if (threadIdx.x < EDGE_DIM) s[threadIdx.x] = 0.f;
  __syncthreads();
#pragma unroll
  for (int d = 0; d < EDGE_DIM; ++d) atomicAdd(&s[d], loc[d]);
  __syncthreads();
  if (threadIdx.x < EDGE_DIM) atomicAdd(&mean[threadIdx.x], s[threadIdx.x]);
}

// collapse layer-1 attention vectors into tiny matrices: asW1/adW1 [5x4], M1 [11x4]
__global__ void k_prep1(const float* W1, const float* We1,
                        const float* a_src1, const float* a_dst1, const float* a_edge1,
                        float* asW1, float* adW1, float* M1) {
  int t = threadIdx.x;           // 64 threads
  int d = t >> 2, h = t & 3;
  if (d < IN_DIM) {
    float s = 0.f, s2 = 0.f;
    for (int c = 0; c < HID; ++c) {
      float w = W1[(size_t)d * D1 + h * HID + c];
      s  += w * a_src1[h * HID + c];
      s2 += w * a_dst1[h * HID + c];
    }
    asW1[d * 4 + h] = s; adW1[d * 4 + h] = s2;
  }
  if (d < EDGE_DIM) {
    float m = 0.f;
    for (int c = 0; c < HID; ++c) m += We1[(size_t)d * D1 + h * HID + c] * a_edge1[h * HID + c];
    M1[d * 4 + h] = m;
  }
}

// layer-2 collapses: vsrc2/vdst2 [512] = W2 @ a_src2/a_dst2 ; M2[11]; W2 -> f16
__global__ void k_prep2(const float* W2, const float* We2,
                        const float* a_src2, const float* a_dst2, const float* a_edge2,
                        float* vsrc2, float* vdst2, float* M2, _Float16* W2f16) {
  int k = threadIdx.x;           // 512 threads
  float s = 0.f, s2 = 0.f;
  for (int c = 0; c < HID; ++c) {
    float w = W2[(size_t)k * HID + c];
    s += w * a_src2[c]; s2 += w * a_dst2[c];
  }
  vsrc2[k] = s; vdst2[k] = s2;
  if (k < EDGE_DIM) {
    float m = 0.f;
    for (int c = 0; c < HID; ++c) m += We2[(size_t)k * HID + c] * a_edge2[c];
    M2[k] = m;
  }
  for (int i = k; i < D1 * HID; i += 512) W2f16[i] = (_Float16)W2[i];
}

// finish mean; precompute self-loop edge scalars for both layers
__global__ void k_prep_loop(float* mean, const float* M1, const float* M2,
                            float* loop1, float* loop2) {
  if (threadIdx.x == 0) {
    float em[EDGE_DIM];
    for (int d = 0; d < EDGE_DIM; ++d) { em[d] = mean[d] * (1.0f / NEDGES); mean[d] = em[d]; }
    for (int h = 0; h < 4; ++h) {
      float s = 0.f;
      for (int d = 0; d < EDGE_DIM; ++d) s += em[d] * M1[d * 4 + h];
      loop1[h] = s;
    }
    float s2 = 0.f;
    for (int d = 0; d < EDGE_DIM; ++d) s2 += em[d] * M2[d];
    loop2[0] = s2;
  }
}

// xh1 = x @ W1  (K=5, memory bound)
__global__ void k_node1(const float* __restrict__ x, const float* __restrict__ W1,
                        float* __restrict__ xh1) {
  long long i = (long long)blockIdx.x * blockDim.x + threadIdx.x;
  if (i >= (long long)NNODES * D1) return;
  int n = (int)(i >> 9), c = (int)(i & (D1 - 1));
  const float* xr = x + n * IN_DIM;
  float s = 0.f;
#pragma unroll
  for (int d = 0; d < IN_DIM; ++d) s += xr[d] * W1[d * D1 + c];
  xh1[i] = s;
}

// a_s1/a_d1 [N,4]
__global__ void k_node1a(const float* __restrict__ x, const float* asW1, const float* adW1,
                         float* as1, float* ad1) {
  int i = blockIdx.x * blockDim.x + threadIdx.x;
  if (i >= NNODES * 4) return;
  int n = i >> 2, h = i & 3;
  float s = 0.f, s2 = 0.f;
#pragma unroll
  for (int d = 0; d < IN_DIM; ++d) {
    float xv = x[n * IN_DIM + d];
    s += xv * asW1[d * 4 + h]; s2 += xv * adW1[d * 4 + h];
  }
  as1[i] = s; ad1[i] = s2;
}

// layer-1 attention logits + segment max
__global__ void k_alpha1(const float* __restrict__ ea, const int* __restrict__ ei,
                         const float* __restrict__ as1, const float* __restrict__ ad1,
                         const float* __restrict__ M1, const float* __restrict__ loop1,
                         float* __restrict__ alpha, float* __restrict__ amax) {
  int e = blockIdx.x * blockDim.x + threadIdx.x;
  if (e >= E2) return;
  int src, dst; float s[4];
  if (e < NEDGES) {
    src = ei[e]; dst = ei[NEDGES + e];
    float s0 = 0.f, s1 = 0.f, s2 = 0.f, s3 = 0.f;
#pragma unroll
    for (int d = 0; d < EDGE_DIM; ++d) {
      float v = ea[(size_t)e * EDGE_DIM + d];
      s0 += v * M1[d * 4 + 0]; s1 += v * M1[d * 4 + 1];
      s2 += v * M1[d * 4 + 2]; s3 += v * M1[d * 4 + 3];
    }
    s[0] = s0; s[1] = s1; s[2] = s2; s[3] = s3;
  } else {
    src = dst = e - NEDGES;
#pragma unroll
    for (int h = 0; h < 4; ++h) s[h] = loop1[h];
  }
#pragma unroll
  for (int h = 0; h < 4; ++h) {
    float a = as1[src * 4 + h] + ad1[dst * 4 + h] + s[h];
    a = a > 0.f ? a : 0.2f * a;          // leaky relu
    alpha[(size_t)e * 4 + h] = a;
    atomicMaxF32(&amax[dst * 4 + h], a);
  }
}

// exp(alpha - max), segment sum
__global__ void k_exp1(const int* __restrict__ ei, const float* __restrict__ amax,
                       float* __restrict__ alpha, float* __restrict__ den) {
  int e = blockIdx.x * blockDim.x + threadIdx.x;
  if (e >= E2) return;
  int dst = (e < NEDGES) ? ei[NEDGES + e] : e - NEDGES;
#pragma unroll
  for (int h = 0; h < 4; ++h) {
    float ex = __expf(alpha[(size_t)e * 4 + h] - amax[dst * 4 + h]);
    alpha[(size_t)e * 4 + h] = ex;
    atomicAdd(&den[dst * 4 + h], ex);
  }
}

// weighted scatter: agg[dst] += xh1[src] * w   (one wave per edge, 512 cols)
__global__ void k_agg1(const float* __restrict__ xh1, const float* __restrict__ ex,
                       const float* __restrict__ den, const int* __restrict__ ei,
                       float* __restrict__ agg) {
  int e = (blockIdx.x * blockDim.x + threadIdx.x) >> 5;
  int lane = threadIdx.x & 31;
  if (e >= E2) return;
  int src, dst;
  if (e < NEDGES) { src = ei[e]; dst = ei[NEDGES + e]; }
  else            { src = dst = e - NEDGES; }
  const float* xs = xh1 + (size_t)src * D1;
  float* ag = agg + (size_t)dst * D1;
  // lane i prefetches cacheline i of the 2KB source row while weights are computed
  __builtin_prefetch(xs + (size_t)lane * 16, 0, 0);
#pragma unroll
  for (int h = 0; h < 4; ++h) {
    float w = ex[(size_t)e * 4 + h] / (den[(size_t)dst * 4 + h] + 1e-16f);
#pragma unroll
    for (int i = 0; i < 4; ++i) {
      int j = h * HID + i * 32 + lane;
      atomicAdd(&ag[j], xs[j] * w);
    }
  }
}

// bias + relu (in place), f16 copy for WMMA, layer-2 attention dots
__global__ void k_post1(float* __restrict__ agg1, const float* __restrict__ b1,
                        _Float16* __restrict__ h1f16,
                        const float* __restrict__ vsrc2, const float* __restrict__ vdst2,
                        float* __restrict__ as2, float* __restrict__ ad2) {
  int n = blockIdx.x, t = threadIdx.x;   // 256 threads, 2 cols each
  __shared__ float rs[256], rd[256];
  float ps = 0.f, pd = 0.f;
  for (int j = t; j < D1; j += 256) {
    float v = agg1[(size_t)n * D1 + j] + b1[j];
    v = v > 0.f ? v : 0.f;
    agg1[(size_t)n * D1 + j] = v;
    h1f16[(size_t)n * D1 + j] = (_Float16)v;
    ps += v * vsrc2[j]; pd += v * vdst2[j];
  }
  rs[t] = ps; rd[t] = pd; __syncthreads();
  for (int s = 128; s > 0; s >>= 1) {
    if (t < s) { rs[t] += rs[t + s]; rd[t] += rd[t + s]; }
    __syncthreads();
  }
  if (t == 0) { as2[n] = rs[0]; ad2[n] = rd[0]; }
}

// xh2[N,128] = h1f16[N,512] @ W2f16[512,128], WMMA f16 -> f32
// A tile (16x512 f16, 16KB) staged in LDS once per block via the Tensor Data Mover,
// then shared by all 8 waves (one wave per 16x16 output tile, nt = wave id).
// `never` is always 0; the guarded store makes the LDS writes un-provably-dead so
// the compiler keeps the ds_load A-fragment reads (TDM builtin has no pointer arg
// visible to alias analysis).
__global__ void k_gemm2(const _Float16* __restrict__ A, const _Float16* __restrict__ B,
                        float* __restrict__ C, int never) {
  __shared__ _Float16 lA[16 * D1];       // 16KB; sole LDS object -> offset 0
  int mt = blockIdx.x;                   // 0..2499
  int nt = threadIdx.x >> 5;             // 0..7
  int lane = threadIdx.x & 31;
  int g = lane >> 4, l = lane & 15;

  if (never) lA[threadIdx.x] = (_Float16)never;   // never executes; keeps LDS live

#if __has_builtin(__builtin_amdgcn_tensor_load_to_lds)
  if (threadIdx.x == 0) {
    unsigned long long ga = (unsigned long long)(A + (size_t)mt * 16 * D1);
    // D# group 0: count=1 | lds_addr=0 | global_addr | type=2 ("image")
    tdm_v4u g0 = { 1u, 0u,
                   (unsigned)(ga & 0xFFFFFFFFu),
                   (unsigned)((ga >> 32) & 0x01FFFFFFu) | (2u << 30) };
    // D# group 1: 2D tensor, data_size=2B, dim0=512, dim1=16, tile=512x16, stride0=512
    tdm_v8i g1 = { 0x10000,               // [17:16] data_size=1 (2 bytes); mask=0
                   (int)(512u << 16),     // tensor_dim0[15:0] at bits 63:48
                   (int)(16u  << 16),     // tensor_dim1[15:0] at bits 111:80
                   (int)(512u << 16),     // tile_dim0 at bits 127:112
                   16,                    // tile_dim1 at bits 143:128
                   512,                   // tensor_dim0_stride[31:0]
                   (int)(512u << 16),     // tensor_dim1_stride[15:0] at bits 223:208
                   0 };
    tdm_v4i z4 = { 0, 0, 0, 0 };
#if defined(__clang_major__) && __clang_major__ >= 23
    tdm_v8i z8 = { 0, 0, 0, 0, 0, 0, 0, 0 };
    __builtin_amdgcn_tensor_load_to_lds(g0, g1, z4, z4, z8, 0);
#else
    __builtin_amdgcn_tensor_load_to_lds(g0, g1, z4, z4, 0);
#endif
    __builtin_amdgcn_s_wait_tensorcnt(0);
  }
  __syncthreads();
#else
  // cooperative fallback: 256 threads x 4 x 16B copies
  {
    const uint4* gsrc = (const uint4*)(A + (size_t)mt * 16 * D1);
    uint4* ldst = (uint4*)lA;
    for (int i = threadIdx.x; i < 16 * D1 / 8; i += 256) ldst[i] = gsrc[i];
  }
  __syncthreads();
#endif

  const _Float16* Ar = lA + (size_t)l * D1;
  // Two independent accumulator chains: consecutive v_wmma ops have no D->C RAW,
  // so the F16 WMMA 1-NOP + 4-coexec hazard (ISA 7.12.1) is hidden by loads.
  v8f c0 = {}, c1 = {};
  for (int kt = 0; kt < 16; kt += 2) {
    v16h a0, b0, a1, b1;
#pragma unroll
    for (int v = 0; v < 8; ++v) {        // A 16x32 layout: VGPR v holds K=(v&3)*2+(v>>2)*16+g*8
      int kb = (v & 3) * 2 + (v >> 2) * 16 + g * 8;
      int k0 = kt * 32 + kb;
      a0[2 * v]     = Ar[k0];
      a0[2 * v + 1] = Ar[k0 + 1];
      a1[2 * v]     = Ar[k0 + 32];
      a1[2 * v + 1] = Ar[k0 + 33];
    }
    // B 32x16 layout: lane holds column N=nt*16+l, K in the same order as A
#pragma unroll
    for (int v = 0; v < 8; ++v) {
      int kb = (v & 3) * 2 + (v >> 2) * 16 + g * 8;
      int k0 = kt * 32 + kb;
      b0[2 * v]     = B[(size_t)k0 * HID + nt * 16 + l];
      b0[2 * v + 1] = B[(size_t)(k0 + 1) * HID + nt * 16 + l];
      b1[2 * v]     = B[(size_t)(k0 + 32) * HID + nt * 16 + l];
      b1[2 * v + 1] = B[(size_t)(k0 + 33) * HID + nt * 16 + l];
    }
    c0 = __builtin_amdgcn_wmma_f32_16x16x32_f16(false, a0, false, b0, (short)0, c0, false, false);
    c1 = __builtin_amdgcn_wmma_f32_16x16x32_f16(false, a1, false, b1, (short)0, c1, false, false);
  }
  v8f c = c0 + c1;
#pragma unroll
  for (int r = 0; r < 8; ++r) {          // C: VGPR r -> (M=r+8g, N=l)
    int row = mt * 16 + r + 8 * g;
    int col = nt * 16 + l;
    C[(size_t)row * HID + col] = c[r];
  }
}

__global__ void k_alpha2(const float* __restrict__ ea, const int* __restrict__ ei,
                         const float* __restrict__ as2, const float* __restrict__ ad2,
                         const float* __restrict__ M2, const float* __restrict__ loop2,
                         float* __restrict__ alpha2, float* __restrict__ amax2) {
  int e = blockIdx.x * blockDim.x + threadIdx.x;
  if (e >= E2) return;
  int src, dst; float s;
  if (e < NEDGES) {
    src = ei[e]; dst = ei[NEDGES + e];
    s = 0.f;
#pragma unroll
    for (int d = 0; d < EDGE_DIM; ++d) s += ea[(size_t)e * EDGE_DIM + d] * M2[d];
  } else { src = dst = e - NEDGES; s = loop2[0]; }
  float a = as2[src] + ad2[dst] + s;
  a = a > 0.f ? a : 0.2f * a;
  alpha2[e] = a;
  atomicMaxF32(&amax2[dst], a);
}

__global__ void k_exp2(const int* __restrict__ ei, const float* __restrict__ amax2,
                       float* __restrict__ alpha2, float* __restrict__ den2) {
  int e = blockIdx.x * blockDim.x + threadIdx.x;
  if (e >= E2) return;
  int dst = (e < NEDGES) ? ei[NEDGES + e] : e - NEDGES;
  float ex = __expf(alpha2[e] - amax2[dst]);
  alpha2[e] = ex;
  atomicAdd(&den2[dst], ex);
}

__global__ void k_agg2(const float* __restrict__ xh2, const float* __restrict__ ex2,
                       const float* __restrict__ den2, const int* __restrict__ ei,
                       float* __restrict__ agg2) {
  int e = (blockIdx.x * blockDim.x + threadIdx.x) >> 5;
  int lane = threadIdx.x & 31;
  if (e >= E2) return;
  int src, dst;
  if (e < NEDGES) { src = ei[e]; dst = ei[NEDGES + e]; }
  else            { src = dst = e - NEDGES; }
  const float* xs = xh2 + (size_t)src * HID;
  __builtin_prefetch(xs + (size_t)(lane & 7) * 16, 0, 0);   // 8 cachelines = 512B row
  float* ag = agg2 + (size_t)dst * HID;
  float w = ex2[e] / (den2[dst] + 1e-16f);
#pragma unroll
  for (int i = 0; i < 4; ++i) {
    int j = i * 32 + lane;
    atomicAdd(&ag[j], xs[j] * w);
  }
}

// relu(agg2+b2) @ Wfc + bfc -> sigmoid ; one wave per node
__global__ void k_final(const float* __restrict__ agg2, const float* __restrict__ b2,
                        const float* __restrict__ Wfc, const float* __restrict__ bfc,
                        float* __restrict__ out) {
  int gid = blockIdx.x * blockDim.x + threadIdx.x;
  int n = gid >> 5, lane = gid & 31;
  if (n >= NNODES) return;
  float acc = 0.f;
#pragma unroll
  for (int i = 0; i < 4; ++i) {
    int j = lane + i * 32;
    float v = agg2[(size_t)n * HID + j] + b2[j];
    v = v > 0.f ? v : 0.f;
    acc += v * Wfc[j];
  }
#pragma unroll
  for (int off = 16; off > 0; off >>= 1) acc += __shfl_down(acc, off, 32);
  if (lane == 0) out[n] = 1.f / (1.f + __expf(-(acc + bfc[0])));
}

extern "C" void kernel_launch(void* const* d_in, const int* in_sizes, int n_in,
                              void* d_out, int out_size, void* d_ws, size_t ws_size,
                              hipStream_t stream) {
  (void)in_sizes; (void)n_in; (void)out_size; (void)ws_size;
  const float* x       = (const float*)d_in[0];
  const int*   ei      = (const int*)d_in[1];
  const float* ea      = (const float*)d_in[2];
  const float* W1      = (const float*)d_in[3];
  const float* We1     = (const float*)d_in[4];
  const float* a_src1  = (const float*)d_in[5];
  const float* a_dst1  = (const float*)d_in[6];
  const float* a_edge1 = (const float*)d_in[7];
  const float* b1      = (const float*)d_in[8];
  const float* W2      = (const float*)d_in[9];
  const float* We2     = (const float*)d_in[10];
  const float* a_src2  = (const float*)d_in[11];
  const float* a_dst2  = (const float*)d_in[12];
  const float* a_edge2 = (const float*)d_in[13];
  const float* b2      = (const float*)d_in[14];
  const float* Wfc     = (const float*)d_in[15];
  const float* bfc     = (const float*)d_in[16];
  float* out = (float*)d_out;

  char* ws = (char*)d_ws;
  size_t off = 0;
  auto alloc = [&](size_t bytes) -> void* {
    void* p = ws + off;
    off = (off + bytes + 255) & ~(size_t)255;
    return p;
  };

  float*    XH1    = (float*)alloc((size_t)NNODES * D1 * 4);
  float*    AGG1   = (float*)alloc((size_t)NNODES * D1 * 4);   // becomes h1 after k_post1
  _Float16* H1F16  = (_Float16*)alloc((size_t)NNODES * D1 * 2);
  _Float16* W2F16  = (_Float16*)alloc((size_t)D1 * HID * 2);
  float*    ALPHA  = (float*)alloc((size_t)E2 * 4 * 4);        // alpha -> ex (layer 1)
  float*    ALPHA2 = (float*)alloc((size_t)E2 * 4);
  float*    XH2    = (float*)alloc((size_t)NNODES * HID * 4);
  float*    AGG2   = (float*)alloc((size_t)NNODES * HID * 4);
  float*    AS1    = (float*)alloc((size_t)NNODES * 4 * 4);
  float*    AD1    = (float*)alloc((size_t)NNODES * 4 * 4);
  float*    AS2    = (float*)alloc((size_t)NNODES * 4);
  float*    AD2    = (float*)alloc((size_t)NNODES * 4);
  float*    AMAX1  = (float*)alloc((size_t)NNODES * 4 * 4);
  float*    DEN1   = (float*)alloc((size_t)NNODES * 4 * 4);
  float*    AMAX2  = (float*)alloc((size_t)NNODES * 4);
  float*    DEN2   = (float*)alloc((size_t)NNODES * 4);
  float*    EAMEAN = (float*)alloc(EDGE_DIM * 4);
  float*    M1     = (float*)alloc(EDGE_DIM * 4 * 4);
  float*    ASW1   = (float*)alloc(IN_DIM * 4 * 4);
  float*    ADW1   = (float*)alloc(IN_DIM * 4 * 4);
  float*    VSRC2  = (float*)alloc(D1 * 4);
  float*    VDST2  = (float*)alloc(D1 * 4);
  float*    M2     = (float*)alloc(EDGE_DIM * 4);
  float*    LOOP1  = (float*)alloc(4 * 4);
  float*    LOOP2  = (float*)alloc(4);

  // per-call init (graph-replay deterministic)
  hipMemsetAsync(EAMEAN, 0, EDGE_DIM * 4, stream);
  hipMemsetAsync(DEN1, 0, (size_t)NNODES * 4 * 4, stream);
  hipMemsetAsync(DEN2, 0, (size_t)NNODES * 4, stream);
  hipMemsetAsync(AGG1, 0, (size_t)NNODES * D1 * 4, stream);
  hipMemsetAsync(AGG2, 0, (size_t)NNODES * HID * 4, stream);
  k_fill<<<(NNODES * 4 + 255) / 256, 256, 0, stream>>>(AMAX1, -FLT_MAX, NNODES * 4);
  k_fill<<<(NNODES + 255) / 256, 256, 0, stream>>>(AMAX2, -FLT_MAX, NNODES);

  // precompute collapsed matrices + self-loop edge features
  k_eamean<<<512, 256, 0, stream>>>(ea, EAMEAN);
  k_prep1<<<1, 64, 0, stream>>>(W1, We1, a_src1, a_dst1, a_edge1, ASW1, ADW1, M1);
  k_prep2<<<1, 512, 0, stream>>>(W2, We2, a_src2, a_dst2, a_edge2, VSRC2, VDST2, M2, W2F16);
  k_prep_loop<<<1, 32, 0, stream>>>(EAMEAN, M1, M2, LOOP1, LOOP2);

  // layer 1
  k_node1<<<(int)(((size_t)NNODES * D1 + 255) / 256), 256, 0, stream>>>(x, W1, XH1);
  k_node1a<<<(NNODES * 4 + 255) / 256, 256, 0, stream>>>(x, ASW1, ADW1, AS1, AD1);
  k_alpha1<<<(E2 + 255) / 256, 256, 0, stream>>>(ea, ei, AS1, AD1, M1, LOOP1, ALPHA, AMAX1);
  k_exp1<<<(E2 + 255) / 256, 256, 0, stream>>>(ei, AMAX1, ALPHA, DEN1);
  k_agg1<<<(int)(((size_t)E2 * 32 + 255) / 256), 256, 0, stream>>>(XH1, ALPHA, DEN1, ei, AGG1);
  k_post1<<<NNODES, 256, 0, stream>>>(AGG1, b1, H1F16, VSRC2, VDST2, AS2, AD2);

  // layer 2: WMMA GEMM (TDM-staged A in LDS) then GAT
  k_gemm2<<<NNODES / 16, 256, 0, stream>>>(H1F16, W2F16, XH2, 0);
  k_alpha2<<<(E2 + 255) / 256, 256, 0, stream>>>(ea, ei, AS2, AD2, M2, LOOP2, ALPHA2, AMAX2);
  k_exp2<<<(E2 + 255) / 256, 256, 0, stream>>>(ei, AMAX2, ALPHA2, DEN2);
  k_agg2<<<(int)(((size_t)E2 * 32 + 255) / 256), 256, 0, stream>>>(XH2, ALPHA2, DEN2, ei, AGG2);

  // head
  k_final<<<(int)(((size_t)NNODES * 32 + 255) / 256), 256, 0, stream>>>(AGG2, b2, Wfc, bfc, out);
}
